// TensorProductRescale_29927332118761
// MI455X (gfx1250) — matrix operand order repaired and checked by reading, not compiled, not run on hardware
//
#include <hip/hip_runtime.h>

typedef __attribute__((ext_vector_type(16))) _Float16 v16h;
typedef __attribute__((ext_vector_type(8)))  float    v8f;

// f16 element offsets of transposed weights inside d_ws
#define OFF000 0        // 256x256
#define OFF110 65536    // 256x128  (w110 is K=128,N=256 -> Wt 256 rows x 128)
#define OFF011 98304    // 128x256
#define OFF101 131072   // 128x128
#define OFF111 147456   // 128x128
#define WS_F16_TOTAL 163840  // 320 KB

__device__ __forceinline__ v16h load_a_frag(const _Float16* base, int stride, int lane, int ks) {
  // 16-bit A layout: lanes 0-15 row M=lane, K = ks*32 + {0..7, 16..23};
  //                  lanes 16-31 row M=lane-16, K = ks*32 + {8..15, 24..31}
  const int row  = lane & 15;
  const int half = lane >> 4;
  const _Float16* p = base + row * stride + ks * 32 + half * 8;
  v16h f;
  ((uint4*)&f)[0] = *(const uint4*)(p);        // K block 0
  ((uint4*)&f)[1] = *(const uint4*)(p + 16);   // K block +16
  return f;
}

__device__ __forceinline__ v16h load_b_frag(const _Float16* wt, int K, int lane, int nb, int ks) {
  // 16-bit B layout: lane n<16 holds col N=n, K = ks*32+{0..7,16..23}; lane n+16 holds K = ks*32+{8..15,24..31}
  // wt is stored transposed (N x K row-major), so each lane reads two 16-byte runs 32 bytes apart.
  const int col   = (lane & 15) + nb * 16;
  const int khalf = lane >> 4;
  const _Float16* p = wt + col * K + ks * 32 + khalf * 8;
  v16h f;
  ((uint4*)&f)[0] = *(const uint4*)(p);
  ((uint4*)&f)[1] = *(const uint4*)(p + 16);
  return f;
}

#define WMMA_F16(a, b, c) \
  __builtin_amdgcn_wmma_f32_16x16x32_f16(false, (a), false, (b), (short)0, (c), false, false)

__global__ __launch_bounds__(256)
void tpr_weight_to_f16t(const float* __restrict__ W, _Float16* __restrict__ Wt, int K, int N) {
  int idx = blockIdx.x * 256 + threadIdx.x;   // idx over K*N, coalesced read of W
  if (idx >= K * N) return;
  int k = idx / N;
  int n = idx - k * N;
  Wt[n * K + k] = (_Float16)W[idx];
}

__global__ __launch_bounds__(256)
void tpr_wmma_kernel(const float* __restrict__ x, const float* __restrict__ y,
                     const float* __restrict__ b0, const _Float16* __restrict__ wt,
                     float* __restrict__ out, int nrows) {
  constexpr float SQK0 = 0.05103103630798288f;   // 1/sqrt(384)
  constexpr float SQK1 = 0.04419417382415922f;   // 1/sqrt(512)
  constexpr float IS3  = 0.5773502691896258f;    // 1/sqrt(3)
  constexpr float IS2  = 0.7071067811865476f;    // 1/sqrt(2)
  constexpr int SA0 = 264;   // padded f16 stride (256 + 8) -> bank-conflict-free row reads
  constexpr int SA1 = 136;   // padded f16 stride (128 + 8)

  __shared__ __align__(16) _Float16 sA0[16 * SA0];      // x0 tile (f16)
  __shared__ __align__(16) _Float16 sA1[16 * SA1];      // x1 . y1
  __shared__ __align__(16) _Float16 sX1[3][16 * SA1];   // x1[:,:,i]
  __shared__ __align__(16) _Float16 sCR[3][16 * SA1];   // cross(x1,y1)[:,:,i]
  __shared__ __align__(16) float    sY[16][4];

  const int  t       = threadIdx.x;
  const long rowBase = (long)blockIdx.x * 16;

  // ---------------- stage A operands into LDS ----------------
  {
    // x0: 16 rows x 256 f32 -> f16; thread t handles (row t/16, cols (t%16)*16..+15)
    const int r = t >> 4;
    const int c = (t & 15) * 16;
    long row = rowBase + r; if (row >= nrows) row = nrows - 1;
    const float* px = x + row * 640 + c;
#pragma unroll
    for (int j = 0; j < 16; j += 4) {
      float4 v = *(const float4*)(px + j);
      sA0[r * SA0 + c + j + 0] = (_Float16)v.x;
      sA0[r * SA0 + c + j + 1] = (_Float16)v.y;
      sA0[r * SA0 + c + j + 2] = (_Float16)v.z;
      sA0[r * SA0 + c + j + 3] = (_Float16)v.w;
    }
  }
  if (t < 16) {
    long row = rowBase + t; if (row >= nrows) row = nrows - 1;
    float4 v = *(const float4*)(y + row * 4);
    sY[t][0] = v.x; sY[t][1] = v.y; sY[t][2] = v.z; sY[t][3] = v.w;
  }
  {
    // x1-derived operands: thread t handles (row t/16, u = (t%16)*8 .. +7)
    const int r  = t >> 4;
    const int u0 = (t & 15) * 8;
    long row = rowBase + r; if (row >= nrows) row = nrows - 1;
    const float* px = x + row * 640 + 256 + 3 * u0;   // x1[row, u0.., :] (24 contiguous f32)
    float xv[24];
#pragma unroll
    for (int j = 0; j < 24; j += 4) *(float4*)(&xv[j]) = *(const float4*)(px + j);
    const float* py = y + row * 4;
    const float ya = py[1], yb = py[2], yc = py[3];
#pragma unroll
    for (int q = 0; q < 8; ++q) {
      const float a = xv[3 * q + 0], b = xv[3 * q + 1], c = xv[3 * q + 2];
      const int u = u0 + q;
      sA1[r * SA1 + u]    = (_Float16)(a * ya + b * yb + c * yc);
      sX1[0][r * SA1 + u] = (_Float16)a;
      sX1[1][r * SA1 + u] = (_Float16)b;
      sX1[2][r * SA1 + u] = (_Float16)c;
      sCR[0][r * SA1 + u] = (_Float16)(b * yc - c * yb);
      sCR[1][r * SA1 + u] = (_Float16)(c * ya - a * yc);
      sCR[2][r * SA1 + u] = (_Float16)(a * yb - b * ya);
    }
  }
  __syncthreads();

  // ---------------- WMMA phase: 8 waves own disjoint output column blocks ----------------
  const int lane   = t & 31;
  const int wv     = t >> 5;
  const int half   = lane >> 4;
  const int nrow16 = lane & 15;

  // y rows needed by this lane's C fragment rows (m = half*8 + v)
  float4 yrow[8];
#pragma unroll
  for (int v = 0; v < 8; ++v) yrow[v] = *(const float4*)(&sY[half * 8 + v][0]);

  const _Float16* wt000 = wt + OFF000;
  const _Float16* wt110 = wt + OFF110;
  const _Float16* wt011 = wt + OFF011;
  const _Float16* wt101 = wt + OFF101;
  const _Float16* wt111 = wt + OFF111;

  // ---- Part 1: out0 columns; wave wv owns 16-col blocks {2wv, 2wv+1} of 256 ----
#pragma unroll
  for (int j = 0; j < 2; ++j) {
    const int nb = 2 * wv + j;
    v8f acc000 = {};
#pragma unroll
    for (int ks = 0; ks < 8; ++ks)
      acc000 = WMMA_F16(load_a_frag(sA0, SA0, lane, ks),
                        load_b_frag(wt000, 256, lane, nb, ks), acc000);
    v8f acc110 = {};
#pragma unroll
    for (int ks = 0; ks < 4; ++ks)
      acc110 = WMMA_F16(load_a_frag(sA1, SA1, lane, ks),
                        load_b_frag(wt110, 128, lane, nb, ks), acc110);

    const int col = nb * 16 + nrow16;
    const float bb = b0[col];
#pragma unroll
    for (int v = 0; v < 8; ++v) {
      const long row = rowBase + half * 8 + v;
      if (row < nrows)
        out[row * 640 + col] = SQK0 * (yrow[v].x * acc000[v] + IS3 * acc110[v]) + bb;
    }
  }

  // ---- Part 2: out1; wave wv owns 16-col block wv of the 128 "u" columns ----
  {
    const int nb = wv;
    v8f a011 = {};
#pragma unroll
    for (int ks = 0; ks < 8; ++ks)
      a011 = WMMA_F16(load_a_frag(sA0, SA0, lane, ks),
                      load_b_frag(wt011, 256, lane, nb, ks), a011);
    v8f a101[3], a111[3];
#pragma unroll
    for (int i = 0; i < 3; ++i) {
      v8f p = {};
#pragma unroll
      for (int ks = 0; ks < 4; ++ks)
        p = WMMA_F16(load_a_frag(sX1[i], SA1, lane, ks),
                     load_b_frag(wt101, 128, lane, nb, ks), p);
      a101[i] = p;
      v8f q = {};
#pragma unroll
      for (int ks = 0; ks < 4; ++ks)
        q = WMMA_F16(load_a_frag(sCR[i], SA1, lane, ks),
                     load_b_frag(wt111, 128, lane, nb, ks), q);
      a111[i] = q;
    }

    const int n = nb * 16 + nrow16;
#pragma unroll
    for (int v = 0; v < 8; ++v) {
      const long row = rowBase + half * 8 + v;
      if (row < nrows) {
        const float y0v = yrow[v].x;
        float* po = out + row * 640 + 256 + 3 * n;   // out1[b, u=n, 0..2]
        po[0] = SQK1 * (a011[v] * yrow[v].y + y0v * a101[0][v] + IS2 * a111[0][v]);
        po[1] = SQK1 * (a011[v] * yrow[v].z + y0v * a101[1][v] + IS2 * a111[1][v]);
        po[2] = SQK1 * (a011[v] * yrow[v].w + y0v * a101[2][v] + IS2 * a111[2][v]);
      }
    }
  }
}

extern "C" void kernel_launch(void* const* d_in, const int* in_sizes, int n_in,
                              void* d_out, int out_size, void* d_ws, size_t ws_size,
                              hipStream_t stream) {
  (void)n_in; (void)out_size; (void)ws_size;
  const float* x    = (const float*)d_in[0];
  const float* y    = (const float*)d_in[1];
  const float* w000 = (const float*)d_in[2];
  const float* w110 = (const float*)d_in[3];
  const float* w011 = (const float*)d_in[4];
  const float* w101 = (const float*)d_in[5];
  const float* w111 = (const float*)d_in[6];
  const float* b0   = (const float*)d_in[7];
  float* out = (float*)d_out;
  _Float16* wt = (_Float16*)d_ws;

  // Convert weights to f16 transposed (N x K) layouts in workspace (deterministic, every call).
  tpr_weight_to_f16t<<<(256 * 256 + 255) / 256, 256, 0, stream>>>(w000, wt + OFF000, 256, 256);
  tpr_weight_to_f16t<<<(128 * 256 + 255) / 256, 256, 0, stream>>>(w110, wt + OFF110, 128, 256);
  tpr_weight_to_f16t<<<(256 * 128 + 255) / 256, 256, 0, stream>>>(w011, wt + OFF011, 256, 128);
  tpr_weight_to_f16t<<<(128 * 128 + 255) / 256, 256, 0, stream>>>(w101, wt + OFF101, 128, 128);
  tpr_weight_to_f16t<<<(128 * 128 + 255) / 256, 256, 0, stream>>>(w111, wt + OFF111, 128, 128);

  const int nrows = in_sizes[0] / 640;                 // B = 150000
  const int nblocks = (nrows + 15) / 16;               // 16 batch rows per workgroup
  tpr_wmma_kernel<<<nblocks, 256, 0, stream>>>(x, y, b0, wt, out, nrows);
}
